// MoEDenseActDense_51969104282036
// MI455X (gfx1250) — matrix operand order repaired and checked by reading, not compiled, not run on hardware
//
#include <hip/hip_runtime.h>

// ---------------------------------------------------------------------------
// MoE dense-act-dense, MI455X (gfx1250, wave32, WMMA bf16, async-to-LDS).
// Forward combine weight is exactly 1 (straight-through gate), so:
//   out[n] = sum over top-4 experts e of  relu(x[n] @ W1[e]^T) @ W2[e]^T
// Pipeline:
//   (1) transpose+cvt W1 -> w1bf[E][D][H] bf16, W2 -> w2bf[E][H][O] bf16
//   (2) cvt x -> xbf[N][D] bf16
//   (3) gate: per-token top-4 bitmask of x@wg
//   (4) GEMM1 (bf16 WMMA): h_all[n][e*H+h] = relu(...) * maskbit  (zeros gate)
//   (5) GEMM2 (bf16 WMMA): out[N][O] = h_all[N][E*H] @ w2bf[E*H][O], K=8192
// A tiles are staged with GLOBAL_LOAD_ASYNC_TO_LDS_B128 (ASYNCcnt) into a
// double-buffered LDS tile; WMMA on buffer t overlaps the async fill of t+1.
// Workspace: xbf 32MB | w1bf 32MB | w2bf 32MB | h_all 128MB | mask 32KB
// ---------------------------------------------------------------------------

typedef __attribute__((ext_vector_type(16))) __bf16 v16bf;
typedef __attribute__((ext_vector_type(8)))  __bf16 v8bf;
typedef __attribute__((ext_vector_type(8)))  float  v8f;

#define N_TOK   8192
#define DIM_D   2048
#define N_EXP   8
#define DIM_H   1024
#define DIM_O   2048
#define KCAT    (N_EXP * DIM_H)   // 8192

#define KC         128            // K-chunk staged in LDS
#define LDS_STRIDE (KC + 8)       // halfs; pad to dodge bank conflicts
#define TILE_HALFS (32 * LDS_STRIDE)

__device__ __forceinline__ unsigned short f32_to_bf16(float f) {
  union { float f; unsigned u; } v; v.f = f;
  unsigned r = v.u + 0x7fffu + ((v.u >> 16) & 1u);   // round-to-nearest-even
  return (unsigned short)(r >> 16);
}

// 32B per thread: global -> LDS direct (ASYNCcnt). INST_OFFSET applies to both
// the LDS and the global address, so one base + offset:16 covers the pair.
__device__ __forceinline__ void async_copy_32b(unsigned lds_off,
                                               const unsigned short* gsrc) {
  unsigned long long ga = (unsigned long long)(uintptr_t)gsrc;
  asm volatile(
      "global_load_async_to_lds_b128 %0, %1, off\n\t"
      "global_load_async_to_lds_b128 %0, %1, off offset:16"
      :: "v"(lds_off), "v"(ga) : "memory");
}

__device__ __forceinline__ void wait_async_all() {
  asm volatile("s_wait_asynccnt 0x0" ::: "memory");
}

// ---------------------------------------------------------------------------
// Wave-level GEMM mainloop. Block tile: 32(M) x 512(N), 8 waves of 32x64.
// A: [32, K] bf16 row-major (global), async-staged into LDS (double buffer).
// Bwave: pre-offset to this wave's n0; [K, ldb] bf16 row-major (global, L2-hot).
// acc[m][n]: 2 x 4 WMMA 16x16 f32 tiles.
// ---------------------------------------------------------------------------
__device__ __forceinline__ void wave_gemm(
    const unsigned short* __restrict__ Ablock, int lda, int K,
    const unsigned short* __restrict__ Bwave, int ldb,
    unsigned short* sA /* __shared__, 2 * TILE_HALFS */, v8f acc[2][4])
{
  const int tid  = threadIdx.x;
  const int lane = tid & 31;
  // A staging: 32 rows x KC halfs; thread -> (row, 32B chunk)
  const int ldRow  = tid >> 3;           // 0..31
  const int ldColH = (tid & 7) * 16;     // halfs, 32B per thread
  // A-fragment addressing (ISA 16-bit A layout: lane%16 = M, lane/16 picks K-half)
  const int fm   = lane & 15;
  const int koff = (lane >> 4) * 8;

  // Flat LDS aperture: low 32 bits of the generic pointer == LDS byte offset.
  const unsigned ldsThr0  = (unsigned)(uintptr_t)sA
                          + (unsigned)(ldRow * LDS_STRIDE + ldColH) * 2u;
  const unsigned bufBytes = (unsigned)TILE_HALFS * 2u;

  const v8f vzero = {0.f, 0.f, 0.f, 0.f, 0.f, 0.f, 0.f, 0.f};
#pragma unroll
  for (int m = 0; m < 2; ++m)
#pragma unroll
    for (int n = 0; n < 4; ++n) acc[m][n] = vzero;

  const unsigned short* aThr = Ablock + (size_t)ldRow * lda + ldColH;
  const int nkt = K / KC;

  // prologue: tile 0 -> buffer 0
  async_copy_32b(ldsThr0, aThr);
  wait_async_all();
  __syncthreads();

  for (int t = 0; t < nkt; ++t) {
    if (t + 1 < nkt)   // fill the other buffer while computing on this one
      async_copy_32b(ldsThr0 + (unsigned)((t + 1) & 1) * bufBytes,
                     aThr + (size_t)(t + 1) * KC);

    const unsigned short* sbuf = sA + (t & 1) * TILE_HALFS;
    const unsigned short* Bk   = Bwave + (size_t)t * KC * ldb;
#pragma unroll
    for (int kk = 0; kk < KC; kk += 32) {
      v16bf afrag[2];
#pragma unroll
      for (int m = 0; m < 2; ++m) {
        const unsigned short* p = sbuf + (m * 16 + fm) * LDS_STRIDE + kk + koff;
        v8bf a0 = *(const v8bf*)(p);        // K = koff+0..7
        v8bf a1 = *(const v8bf*)(p + 16);   // K = koff+16..23
        afrag[m] = __builtin_shufflevector(a0, a1,
            0, 1, 2, 3, 4, 5, 6, 7, 8, 9, 10, 11, 12, 13, 14, 15);
      }
#pragma unroll
      for (int n = 0; n < 4; ++n) {
        // B fragment: lane%16 = N column, elements follow A's K order
        v16bf bfrag;
#pragma unroll
        for (int s = 0; s < 16; ++s)
          bfrag[s] = ((const __bf16*)Bk)[(size_t)(kk + koff + (s & 7) + (s >> 3) * 16) * ldb + n * 16 + fm];
#pragma unroll
        for (int m = 0; m < 2; ++m)
          acc[m][n] = __builtin_amdgcn_wmma_f32_16x16x32_bf16(
              false, afrag[m], false, bfrag, (short)0, acc[m][n], false, false);
      }
    }
    wait_async_all();     // next buffer resident before anyone reads it
    __syncthreads();      // and all waves done with the buffer we overwrite next
  }
}

// ---------------------------------------------------------------------------
// (1) Transpose + f32->bf16: in [E][R][C] -> out [E][C][R]
// ---------------------------------------------------------------------------
__global__ __launch_bounds__(256) void transpose_cvt_kernel(
    const float* __restrict__ in, unsigned short* __restrict__ out, int R, int C)
{
  __shared__ float tile[32][33];
  const int e  = blockIdx.z;
  const float*          inE  = in  + (size_t)e * R * C;
  unsigned short*       outE = out + (size_t)e * R * C;
  const int c0 = blockIdx.x * 32, r0 = blockIdx.y * 32;
#pragma unroll
  for (int k = 0; k < 4; ++k)
    tile[threadIdx.y + 8 * k][threadIdx.x] =
        inE[(size_t)(r0 + threadIdx.y + 8 * k) * C + c0 + threadIdx.x];
  __syncthreads();
#pragma unroll
  for (int k = 0; k < 4; ++k)
    outE[(size_t)(c0 + threadIdx.y + 8 * k) * R + r0 + threadIdx.x] =
        f32_to_bf16(tile[threadIdx.x][threadIdx.y + 8 * k]);
}

// ---------------------------------------------------------------------------
// (2) x f32 -> bf16, packed pair stores
// ---------------------------------------------------------------------------
__global__ __launch_bounds__(256) void cvt_x_kernel(
    const float* __restrict__ in, unsigned short* __restrict__ out, size_t npairs)
{
  size_t i = (size_t)blockIdx.x * blockDim.x + threadIdx.x;
  const size_t stride = (size_t)gridDim.x * blockDim.x;
  for (; i < npairs; i += stride) {
    float2 f = ((const float2*)in)[i];
    unsigned lo = f32_to_bf16(f.x), hi = f32_to_bf16(f.y);
    ((unsigned*)out)[i] = lo | (hi << 16);
  }
}

// ---------------------------------------------------------------------------
// (3) Gate: one wave per token; 8 logits, shuffle-reduce, stable top-4 bitmask
// ---------------------------------------------------------------------------
__global__ __launch_bounds__(256) void gate_topk_kernel(
    const float* __restrict__ x, const float* __restrict__ wg,
    unsigned int* __restrict__ mask)
{
  const int token = (blockIdx.x * blockDim.x + threadIdx.x) >> 5;
  const int lane  = threadIdx.x & 31;
  if (token >= N_TOK) return;
  const float* xr = x + (size_t)token * DIM_D;
  float g[N_EXP];
#pragma unroll
  for (int e = 0; e < N_EXP; ++e) g[e] = 0.f;
  for (int d = lane; d < DIM_D; d += 32) {
    const float xv = xr[d];
    const float* wr = wg + (size_t)d * N_EXP;
#pragma unroll
    for (int e = 0; e < N_EXP; ++e) g[e] += xv * wr[e];
  }
#pragma unroll
  for (int off = 16; off > 0; off >>= 1)
#pragma unroll
    for (int e = 0; e < N_EXP; ++e) g[e] += __shfl_xor(g[e], off, 32);
  if (lane == 0) {
    unsigned m = 0;
    for (int k = 0; k < 4; ++k) {          // lax.top_k: lower index wins ties
      int best = 0; float bv = -__builtin_inff();
      for (int e = 0; e < N_EXP; ++e)
        if (!((m >> e) & 1u) && g[e] > bv) { bv = g[e]; best = e; }
      m |= 1u << best;
    }
    mask[token] = m;
  }
}

// ---------------------------------------------------------------------------
// (4) GEMM1: h_all[tok][e*H + h] = relu(x @ w1bf[e]) * maskbit, bf16 out
//     grid: (tokTiles=256, Htiles=2, E=8)
// ---------------------------------------------------------------------------
__global__ __launch_bounds__(256) void gemm1_kernel(
    const unsigned short* __restrict__ xbf,
    const unsigned short* __restrict__ w1bf,
    const unsigned int*   __restrict__ mask,
    unsigned short*       __restrict__ hall)
{
  __shared__ unsigned short sA[2 * TILE_HALFS];
  const int tokT = blockIdx.x, ht = blockIdx.y, e = blockIdx.z;
  const int wave = threadIdx.x >> 5, lane = threadIdx.x & 31;
  const int n0 = ht * 512 + wave * 64;

  const unsigned short* A = xbf + (size_t)tokT * 32 * DIM_D;
  const unsigned short* B = w1bf + (size_t)e * DIM_D * DIM_H + n0;
  v8f acc[2][4];
  wave_gemm(A, DIM_D, DIM_D, B, DIM_H, sA, acc);

  const int col = lane & 15, rbase = (lane >> 4) * 8;
#pragma unroll
  for (int m = 0; m < 2; ++m)
#pragma unroll
    for (int n = 0; n < 4; ++n) {
      const int hcol = n0 + n * 16 + col;
#pragma unroll
      for (int v = 0; v < 8; ++v) {
        const int token = tokT * 32 + m * 16 + rbase + v;
        const float bit = (float)((mask[token] >> e) & 1u);
        const float val = fmaxf(acc[m][n][v], 0.f) * bit;
        hall[(size_t)token * KCAT + e * DIM_H + hcol] = f32_to_bf16(val);
      }
    }
}

// ---------------------------------------------------------------------------
// (5) GEMM2: out[N][O] = h_all[N][8192] @ w2bf[8192][O], f32 out, K=8192
//     grid: (tokTiles=256, Otiles=4)
// ---------------------------------------------------------------------------
__global__ __launch_bounds__(256) void gemm2_kernel(
    const unsigned short* __restrict__ hall,
    const unsigned short* __restrict__ w2bf,
    float*                __restrict__ out)
{
  __shared__ unsigned short sA[2 * TILE_HALFS];
  const int tokT = blockIdx.x, ot = blockIdx.y;
  const int wave = threadIdx.x >> 5, lane = threadIdx.x & 31;
  const int n0 = ot * 512 + wave * 64;

  const unsigned short* A = hall + (size_t)tokT * 32 * KCAT;
  const unsigned short* B = w2bf + n0;
  v8f acc[2][4];
  wave_gemm(A, KCAT, KCAT, B, DIM_O, sA, acc);

  const int col = lane & 15, rbase = (lane >> 4) * 8;
#pragma unroll
  for (int m = 0; m < 2; ++m)
#pragma unroll
    for (int n = 0; n < 4; ++n) {
      const int ocol = n0 + n * 16 + col;
#pragma unroll
      for (int v = 0; v < 8; ++v) {
        const int token = tokT * 32 + m * 16 + rbase + v;
        out[(size_t)token * DIM_O + ocol] = acc[m][n][v];
      }
    }
}

// ---------------------------------------------------------------------------
extern "C" void kernel_launch(void* const* d_in, const int* in_sizes, int n_in,
                              void* d_out, int out_size, void* d_ws, size_t ws_size,
                              hipStream_t stream) {
  const float* x  = (const float*)d_in[0];   // [4,2048,2048]
  const float* wg = (const float*)d_in[1];   // [2048,8]
  const float* w1 = (const float*)d_in[2];   // [8,1024,2048]
  const float* w2 = (const float*)d_in[3];   // [8,2048,1024]
  float* out = (float*)d_out;                // [4,2048,2048]

  char* ws = (char*)d_ws;
  unsigned short* xbf  = (unsigned short*)(ws);                         // 32MB
  unsigned short* w1bf = (unsigned short*)(ws + (size_t)32  * 1048576); // 32MB  [E][D][H]
  unsigned short* w2bf = (unsigned short*)(ws + (size_t)64  * 1048576); // 32MB  [E][H][O]
  unsigned short* hall = (unsigned short*)(ws + (size_t)96  * 1048576); // 128MB [N][E*H]
  unsigned int*   mask = (unsigned int*)  (ws + (size_t)224 * 1048576); // 32KB

  // (1) weight transpose + bf16 cvt (w1: R=H,C=D ; w2: R=O,C=H)
  transpose_cvt_kernel<<<dim3(DIM_D / 32, DIM_H / 32, N_EXP), dim3(32, 8), 0, stream>>>(
      w1, w1bf, DIM_H, DIM_D);
  transpose_cvt_kernel<<<dim3(DIM_H / 32, DIM_O / 32, N_EXP), dim3(32, 8), 0, stream>>>(
      w2, w2bf, DIM_O, DIM_H);
  // (2) x -> bf16
  cvt_x_kernel<<<4096, 256, 0, stream>>>(x, xbf, (size_t)N_TOK * DIM_D / 2);
  // (3) gate top-4 bitmask (one wave per token)
  gate_topk_kernel<<<N_TOK / 8, 256, 0, stream>>>(x, wg, mask);
  // (4) GEMM1 + relu + mask
  gemm1_kernel<<<dim3(N_TOK / 32, DIM_H / 512, N_EXP), 256, 0, stream>>>(
      xbf, w1bf, mask, hall);
  // (5) GEMM2, K = E*H
  gemm2_kernel<<<dim3(N_TOK / 32, DIM_O / 512), 256, 0, stream>>>(hall, w2bf, out);
}
